// MultiHeadAttention_14998025798438
// MI455X (gfx1250) — matrix operand order repaired and checked by reading, not compiled, not run on hardware
//
#include <hip/hip_runtime.h>

#define T_ 512
#define S_ 1024
#define B_ 16
#define A_ 512
#define H_ 8
#define D_ 64
#define NEG_BIG (-3.0e38f)

typedef __attribute__((ext_vector_type(16))) __bf16          v16bf;
typedef __attribute__((ext_vector_type(8)))  __bf16          v8bf;
typedef __attribute__((ext_vector_type(2)))  __bf16          v2bf;
typedef __attribute__((ext_vector_type(8)))  float           v8f;
typedef __attribute__((ext_vector_type(4)))  unsigned        u32x4;
typedef __attribute__((ext_vector_type(8)))  unsigned        u32x8;

// ---------- helpers ----------
// pack two f32 -> packed 2xbf16 dword (native v_cvt if the target has it)
__device__ __forceinline__ unsigned pk_bf16(float a, float b) {
  v2bf p = { (__bf16)a, (__bf16)b };
  return __builtin_bit_cast(unsigned, p);
}
__device__ __forceinline__ unsigned short f2bfu1(float f) {
  return __builtin_bit_cast(unsigned short, (__bf16)f);
}
// concatenate two 8-element bf16 halves as dword moves
__device__ __forceinline__ v16bf cat8(v8bf a, v8bf b) {
  u32x4 al = __builtin_bit_cast(u32x4, a);
  u32x4 bl = __builtin_bit_cast(u32x4, b);
  u32x8 r;
#pragma unroll
  for (int i = 0; i < 4; ++i) { r[i] = al[i]; r[i + 4] = bl[i]; }
  return __builtin_bit_cast(v16bf, r);
}
__device__ __forceinline__ v8f wmma_bf16(v16bf a, v16bf b, v8f c) {
  // D = A(16x32 bf16) x B(32x16 bf16) + C(16x16 f32)
  return __builtin_amdgcn_wmma_f32_16x16x32_bf16(false, a, false, b, (short)0, c,
                                                 false, false);
}

// ---------- prep: f32 [TT,B,A] -> bf16 [B,H,TT,64] ----------
__global__ __launch_bounds__(256) void prep_qk_kernel(const float* __restrict__ src,
                                                      unsigned short* __restrict__ dst,
                                                      int TT) {
  unsigned i = blockIdx.x * 256u + threadIdx.x;   // TT*B*A/8 threads
  int d8 = i & 7;
  unsigned r = i >> 3;
  int t = r % TT;
  unsigned r2 = r / TT;
  int h = r2 % H_;
  int b = r2 / H_;
  const float* s = src + ((size_t)t * B_ + b) * A_ + h * D_ + d8 * 8;
  u32x4 o;
#pragma unroll
  for (int j = 0; j < 4; ++j) o[j] = pk_bf16(s[2 * j], s[2 * j + 1]);
  *(u32x4*)(dst + (((size_t)(b * H_ + h) * TT) + t) * D_ + d8 * 8) = o;
}

// ---------- prep: o_w f32 [A,A] -> bf16 [A,A] ----------
__global__ __launch_bounds__(256) void prep_ow_kernel(const float* __restrict__ ow,
                                                      unsigned short* __restrict__ dst) {
  unsigned i = blockIdx.x * 256u + threadIdx.x;   // A*A/8 threads
  const float* s = ow + (size_t)i * 8;
  u32x4 o;
#pragma unroll
  for (int j = 0; j < 4; ++j) o[j] = pk_bf16(s[2 * j], s[2 * j + 1]);
  *(u32x4*)(dst + (size_t)i * 8) = o;
}

// ---------- v-proj: vtb[b,h,d,s] = bf16( v_w @ kh^T + v_b ) ----------
__global__ __launch_bounds__(256) void vproj_kernel(const float* __restrict__ keys,
                                                    const float* __restrict__ v_w,
                                                    const float* __restrict__ v_b,
                                                    unsigned short* __restrict__ vtb) {
  int lane = threadIdx.x & 31, wv = threadIdx.x >> 5;
  int w = blockIdx.x * 8 + wv;            // 8192 waves = B*H*(S/16)
  int sj = w & 63;
  int bh = w >> 6;
  int b = bh >> 3, h = bh & 7;
  int s0 = sj * 16, lm = lane & 15, hi = lane >> 4;

  // B operand: B[k=d'][n=s] = keys[s, b, h*64 + d'];  n = lm, K(e) = k0 + 16*hi + e
  const float* krow = keys + ((size_t)(s0 + lm) * B_ + b) * A_ + h * D_;
  u32x8 w0, w1;
#pragma unroll
  for (int i = 0; i < 8; ++i) {
    w0[i] = pk_bf16(krow[16 * hi + 2 * i], krow[16 * hi + 2 * i + 1]);
    w1[i] = pk_bf16(krow[32 + 16 * hi + 2 * i], krow[32 + 16 * hi + 2 * i + 1]);
  }
  v16bf bop0 = __builtin_bit_cast(v16bf, w0);
  v16bf bop1 = __builtin_bit_cast(v16bf, w1);

#pragma unroll
  for (int mm = 0; mm < 4; ++mm) {
    // A operand: rows d = mm*16 + lm of v_w;  K(e) = 16*hi + e
    const float* wrow = v_w + (size_t)(mm * 16 + lm) * D_;
    u32x8 wa0, wa1;
#pragma unroll
    for (int i = 0; i < 8; ++i) {
      int k = 16 * hi + 2 * i;
      wa0[i] = pk_bf16(wrow[k], wrow[k + 1]);
      wa1[i] = pk_bf16(wrow[32 + k], wrow[32 + k + 1]);
    }
    v8f c = {};
    c = wmma_bf16(__builtin_bit_cast(v16bf, wa0), bop0, c);
    c = wmma_bf16(__builtin_bit_cast(v16bf, wa1), bop1, c);
    // C layout: M = d = mm*16 + r + 8*hi, N = s = s0 + lm
#pragma unroll
    for (int r = 0; r < 8; ++r) {
      int d = mm * 16 + r + 8 * hi;
      float v = c[r] + v_b[d];
      vtb[((size_t)(b * H_ + h) * D_ + d) * S_ + s0 + lm] = f2bfu1(v);
    }
  }
}

// one 16(s)x16(t) scores^T tile: A = K-tile (M=s, K=d), B = Q^T (K=d, N=t)
__device__ __forceinline__ v8f qk_tile(const unsigned short* krow, int hi,
                                       v16bf qop0, v16bf qop1) {
  v8bf a0lo = *(const v8bf*)(krow + 16 * hi);
  v8bf a0hi = *(const v8bf*)(krow + 16 * hi + 8);
  v8bf a1lo = *(const v8bf*)(krow + 32 + 16 * hi);
  v8bf a1hi = *(const v8bf*)(krow + 32 + 16 * hi + 8);
  v8f sc = {};
  sc = wmma_bf16(cat8(a0lo, a0hi), qop0, sc);
  sc = wmma_bf16(cat8(a1lo, a1hi), qop1, sc);
  return sc;
}

// ---------- flash attention core ----------
__global__ __launch_bounds__(256) void attn_kernel(const unsigned short* __restrict__ qbf,
                                                   const unsigned short* __restrict__ kbf,
                                                   const unsigned short* __restrict__ vtb,
                                                   const unsigned char* __restrict__ mask,
                                                   unsigned short* __restrict__ vals) {
  int lane = threadIdx.x & 31, wv = threadIdx.x >> 5;
  int w = blockIdx.x * 8 + wv;            // 4096 waves = B*H*(T/16)
  int tt = w & 31;
  int bh = w >> 5;
  int b = bh >> 3, h = bh & 7;
  int t0 = tt * 16, lm = lane & 15, hi = lane >> 4;

  // Q B-operand: n = t = lm, K(e) = kd0 + 16*hi + e  (contiguous in d)
  const unsigned short* qrow = qbf + (((size_t)(b * H_ + h) * T_) + (t0 + lm)) * D_;
  v16bf qop0 = *(const v16bf*)(qrow + 16 * hi);
  v16bf qop1 = *(const v16bf*)(qrow + 32 + 16 * hi);

  const unsigned short* kbase = kbf + ((size_t)(b * H_ + h) * S_) * D_;
  const unsigned short* vbase = vtb + ((size_t)(b * H_ + h) * D_) * S_;
  const unsigned char* mrow = mask + ((size_t)b * T_ + (t0 + lm)) * S_;  // row of this lane's t

  float m_r = NEG_BIG, l_r = 0.0f;
  v8f acc[4] = {};

  for (int s0 = 0; s0 < S_; s0 += 32) {
    v8f st0 = qk_tile(kbase + (size_t)(s0 + lm) * D_, hi, qop0, qop1);
    v8f st1 = qk_tile(kbase + (size_t)(s0 + 16 + lm) * D_, hi, qop0, qop1);
    // C layout of scores^T: value[s = 16j + r + 8*hi][t = lm]
    unsigned long long mb0 = *(const unsigned long long*)(mrow + s0 + 8 * hi);
    unsigned long long mb1 = *(const unsigned long long*)(mrow + s0 + 16 + 8 * hi);
    float c0[8], c1[8];
#pragma unroll
    for (int r = 0; r < 8; ++r) {
      float x0 = st0[r] * 0.125f;
      float x1 = st1[r] * 0.125f;
      if (((mb0 >> (8 * r)) & 0xFFull) == 0) x0 = NEG_BIG;
      if (((mb1 >> (8 * r)) & 0xFFull) == 0) x1 = NEG_BIG;
      c0[r] = x0;
      c1[r] = x1;
    }
    // online softmax stats (row = t = lm; combine the two s-halves via xor-16)
    float tmax = c0[0];
#pragma unroll
    for (int r = 0; r < 8; ++r) { tmax = fmaxf(tmax, c0[r]); tmax = fmaxf(tmax, c1[r]); }
    tmax = fmaxf(tmax, __shfl_xor(tmax, 16, 32));
    float mnew = fmaxf(m_r, tmax);
    float corr = __expf(m_r - mnew);
    m_r = mnew;
    float psum = 0.0f;
#pragma unroll
    for (int r = 0; r < 8; ++r) {
      c0[r] = __expf(c0[r] - mnew); psum += c0[r];
      c1[r] = __expf(c1[r] - mnew); psum += c1[r];
    }
    psum += __shfl_xor(psum, 16, 32);
    l_r = l_r * corr + psum;
#pragma unroll
    for (int dd = 0; dd < 4; ++dd)
#pragma unroll
      for (int r = 0; r < 8; ++r) acc[dd][r] *= corr;

    // Build P^T B-operand: n = t = lm, K = s_local = 16*hi + e
    float x[16];
#pragma unroll
    for (int e = 0; e < 8; ++e) {
      float sw = __shfl_xor(c1[e], 16, 32);   // hi lanes receive lo's tile1 s=e
      x[e] = hi ? sw : c0[e];
    }
#pragma unroll
    for (int e = 0; e < 8; ++e) {
      float sw = __shfl_xor(c0[e], 16, 32);   // lo lanes receive hi's tile0 s=8+e
      x[8 + e] = hi ? c1[e] : sw;
    }
    u32x8 pw;
#pragma unroll
    for (int i = 0; i < 8; ++i) pw[i] = pk_bf16(x[2 * i], x[2 * i + 1]);
    v16bf pop = __builtin_bit_cast(v16bf, pw);

    // (PV)^T += V^T x P^T : A = V^T (M = d-chunk, K = s)
#pragma unroll
    for (int dd = 0; dd < 4; ++dd) {
      const unsigned short* vrow = vbase + (size_t)(dd * 16 + lm) * S_ + s0;
      v8bf vlo = *(const v8bf*)(vrow + 16 * hi);
      v8bf vhi = *(const v8bf*)(vrow + 16 * hi + 8);
      acc[dd] = wmma_bf16(cat8(vlo, vhi), pop, acc[dd]);
    }
  }

  float inv = 1.0f / l_r;   // per-lane row sum (t = lm), identical on both halves
#pragma unroll
  for (int dd = 0; dd < 4; ++dd) {
    u32x4 ov;
#pragma unroll
    for (int i = 0; i < 4; ++i)
      ov[i] = pk_bf16(acc[dd][2 * i] * inv, acc[dd][2 * i + 1] * inv);
    // acc C layout: M = d = dd*16 + r + 8*hi, N = t = lm ; vals is [T,B,A] bf16
    *(u32x4*)(vals + ((size_t)(t0 + lm) * B_ + b) * A_ + h * D_ + dd * 16 + 8 * hi) = ov;
  }
}

// ---------- o-proj: out[T*B, A] = vals_bf16 @ o_w^T + o_b (f32 out) ----------
__global__ __launch_bounds__(256) void oproj_kernel(const unsigned short* __restrict__ vals,
                                                    const unsigned short* __restrict__ owbf,
                                                    const float* __restrict__ o_b,
                                                    float* __restrict__ out) {
  int lane = threadIdx.x & 31, wv = threadIdx.x >> 5;
  int w = blockIdx.x * 8 + wv;            // 4096 waves = (T*B/16) * (A/64)
  int nb = w & 7;
  int mb = w >> 3;
  int m0 = mb * 16, n0 = nb * 64;
  int lm = lane & 15, hi = lane >> 4;

  v8f acc[4] = {};
  for (int k0 = 0; k0 < A_; k0 += 32) {
    const unsigned short* arow = vals + (size_t)(m0 + lm) * A_ + k0;
    v8bf alo = *(const v8bf*)(arow + 16 * hi);
    v8bf ahi = *(const v8bf*)(arow + 16 * hi + 8);
    v16bf aop = cat8(alo, ahi);
#pragma unroll
    for (int nn = 0; nn < 4; ++nn) {
      const unsigned short* brow = owbf + (size_t)(n0 + nn * 16 + lm) * A_ + k0 + 16 * hi;
      v16bf bop = *(const v16bf*)brow;
      acc[nn] = wmma_bf16(aop, bop, acc[nn]);
    }
  }
#pragma unroll
  for (int nn = 0; nn < 4; ++nn) {
    int n = n0 + nn * 16 + lm;
    float bias = o_b[n];
#pragma unroll
    for (int r = 0; r < 8; ++r) {
      int row = m0 + r + 8 * hi;
      out[(size_t)row * A_ + n] = acc[nn][r] + bias;
    }
  }
}

extern "C" void kernel_launch(void* const* d_in, const int* in_sizes, int n_in,
                              void* d_out, int out_size, void* d_ws, size_t ws_size,
                              hipStream_t stream) {
  const float* queries = (const float*)d_in[0];
  const float* keys = (const float*)d_in[1];
  const unsigned char* mask = (const unsigned char*)d_in[2];
  const float* v_w = (const float*)d_in[3];
  const float* v_b = (const float*)d_in[4];
  const float* o_w = (const float*)d_in[5];
  const float* o_b = (const float*)d_in[6];
  float* out = (float*)d_out;

  char* ws = (char*)d_ws;
  unsigned short* qbf  = (unsigned short*)(ws);                 //  8 MB  [B,H,T,64]
  unsigned short* kbf  = (unsigned short*)(ws + (8u << 20));    // 16 MB  [B,H,S,64]
  unsigned short* vtb  = (unsigned short*)(ws + (24u << 20));   // 16 MB  [B,H,64,S]
  unsigned short* vals = (unsigned short*)(ws + (40u << 20));   //  8 MB  [T,B,A]
  unsigned short* owbf = (unsigned short*)(ws + (48u << 20));   // 0.5 MB [A,A]

  prep_qk_kernel<<<(T_ * B_ * A_ / 8) / 256, 256, 0, stream>>>(queries, qbf, T_);
  prep_qk_kernel<<<(S_ * B_ * A_ / 8) / 256, 256, 0, stream>>>(keys, kbf, S_);
  prep_ow_kernel<<<(A_ * A_ / 8) / 256, 256, 0, stream>>>(o_w, owbf);
  vproj_kernel<<<(B_ * H_ * (S_ / 16)) / 8, 256, 0, stream>>>(keys, v_w, v_b, vtb);
  attn_kernel<<<(B_ * H_ * (T_ / 16)) / 8, 256, 0, stream>>>(qbf, kbf, vtb, mask, vals);
  oproj_kernel<<<((T_ * B_ / 16) * (A_ / 64)) / 8, 256, 0, stream>>>(vals, owbf, o_b, out);
}
